// VanillaRNN_11046655885450
// MI455X (gfx1250) — matrix operand order repaired and checked by reading, not compiled, not run on hardware
//
#include <hip/hip_runtime.h>
#include <math.h>

typedef __attribute__((ext_vector_type(16))) __bf16          v16bf;
typedef __attribute__((ext_vector_type(16))) unsigned short  v16u;
typedef __attribute__((ext_vector_type(8)))  float           v8f;
typedef __attribute__((ext_vector_type(4)))  int             v4i;

#define TILE_M 128
#define TILE_N 128
#define TILE_K 32
#define LDSK   40      // row stride in halves: 80 B = 5*16 (b128-aligned), 20 dwords -> conflict-free
#define THREADS 256

#define AS1 __attribute__((address_space(1)))
#define AS3 __attribute__((address_space(3)))

#if defined(__HIP_DEVICE_COMPILE__) && __has_builtin(__builtin_amdgcn_global_load_async_to_lds_b128)
#define HAVE_ASYNC 1
#else
#define HAVE_ASYNC 0
#endif

#if HAVE_ASYNC
#if __has_builtin(__builtin_amdgcn_s_wait_asynccnt)
#define WAIT_ASYNC(n) __builtin_amdgcn_s_wait_asynccnt(n)
#else
#define WAIT_ASYNC(n) asm volatile("s_wait_asynccnt %0" ::"i"(n) : "memory")
#endif
#else
#define WAIT_ASYNC(n)
#endif

__device__ __host__ __forceinline__ unsigned short f32_to_bf16(float f) {
    unsigned int u = __builtin_bit_cast(unsigned int, f);
    u += 0x7FFFu + ((u >> 16) & 1u);          // round-to-nearest-even
    return (unsigned short)(u >> 16);
}

// Double-buffered async-DMA WMMA core: acc[0..7] = Abase(128xK) @ Bbase(128xK)^T
// A : rows of M-stripe, bf16, lda==K.  B : rows of N-stripe (pre-transposed), bf16.
__device__ __forceinline__ void mm_core(
    const unsigned short* __restrict__ Abase,
    const unsigned short* __restrict__ Bbase,
    int K,
    unsigned short (*lA)[TILE_M][LDSK],
    unsigned short (*lB)[TILE_N][LDSK],
    int tid, int wave, int hl, int ln,
    v8f acc[8])
{
    // per-thread staging coordinates: 512 16B chunks per tile, 2 per thread
    const int r0 = tid >> 2,             q0 = tid & 3;
    const int r1 = (tid + THREADS) >> 2, q1 = (tid + THREADS) & 3;

    auto stage = [&](int kc, int buf) {
#if HAVE_ASYNC
        __builtin_amdgcn_global_load_async_to_lds_b128(
            (AS1 v4i*)(Abase + (size_t)r0 * K + kc + q0 * 8),
            (AS3 v4i*)&lA[buf][r0][q0 * 8], 0, 0);
        __builtin_amdgcn_global_load_async_to_lds_b128(
            (AS1 v4i*)(Bbase + (size_t)r0 * K + kc + q0 * 8),
            (AS3 v4i*)&lB[buf][r0][q0 * 8], 0, 0);
        __builtin_amdgcn_global_load_async_to_lds_b128(
            (AS1 v4i*)(Abase + (size_t)r1 * K + kc + q1 * 8),
            (AS3 v4i*)&lA[buf][r1][q1 * 8], 0, 0);
        __builtin_amdgcn_global_load_async_to_lds_b128(
            (AS1 v4i*)(Bbase + (size_t)r1 * K + kc + q1 * 8),
            (AS3 v4i*)&lB[buf][r1][q1 * 8], 0, 0);
#else
        *(uint4*)&lA[buf][r0][q0 * 8] = *(const uint4*)(Abase + (size_t)r0 * K + kc + q0 * 8);
        *(uint4*)&lB[buf][r0][q0 * 8] = *(const uint4*)(Bbase + (size_t)r0 * K + kc + q0 * 8);
        *(uint4*)&lA[buf][r1][q1 * 8] = *(const uint4*)(Abase + (size_t)r1 * K + kc + q1 * 8);
        *(uint4*)&lB[buf][r1][q1 * 8] = *(const uint4*)(Bbase + (size_t)r1 * K + kc + q1 * 8);
#endif
    };

    #pragma unroll
    for (int i = 0; i < 8; ++i) {
        #pragma unroll
        for (int j = 0; j < 8; ++j) acc[i][j] = 0.0f;
    }

    const int nIters = K / TILE_K;
    stage(0, 0);                                   // prologue: tile 0 in flight

    for (int it = 0; it < nIters; ++it) {
        const int buf = it & 1;
        if (it + 1 < nIters) {
            stage((it + 1) * TILE_K, buf ^ 1);     // overlap next tile's DMA
            WAIT_ASYNC(4);                         // in-order: first 4 = current tile done
        } else {
            WAIT_ASYNC(0);
        }
        __syncthreads();                           // tile `buf` visible to all waves

        // A fragment: 16x32, rows wave*16..+15.
        //   lanes 0-15: K = e (e<8) / 16+e-8 (e>=8);  lanes 16-31: +8
        v16u au;
        #pragma unroll
        for (int e = 0; e < 16; ++e) {
            int k = (e < 8) ? (hl * 8 + e) : (8 + hl * 8 + e);
            au[e] = lA[buf][wave * 16 + ln][k];
        }
        v16bf afrag = __builtin_bit_cast(v16bf, au);

        v16bf bfrag[8];
        #pragma unroll
        for (int nt = 0; nt < 8; ++nt) {
            v16u bu;
            #pragma unroll
            for (int e = 0; e < 16; ++e)
                bu[e] = lB[buf][nt * 16 + ln][(e < 8) ? (hl * 8 + e) : (8 + hl * 8 + e)];   // N = ln, K = table order (same as A)
            bfrag[nt] = __builtin_bit_cast(v16bf, bu);
        }

        #pragma unroll
        for (int nt = 0; nt < 8; ++nt)
            acc[nt] = __builtin_amdgcn_wmma_f32_16x16x32_bf16(
                false, afrag, false, bfrag[nt], (short)0, acc[nt], false, false);

        __syncthreads();                           // all reads of `buf` done before overwrite
    }
}

// Persistent RNN: entire 256-step recurrence in one kernel.
// Dependence is stripe-local: step t+1's row-stripe m needs only step t's row-stripe m,
// written by the 8 blocks sharing blockIdx.y -> 8-block atomic barrier per stripe per step.
__global__ __launch_bounds__(THREADS)
void rnn_persistent(const unsigned short* __restrict__ WhhT,  // H x H bf16, n-major
                    unsigned short* __restrict__ hA,          // ping (t even -> read)
                    unsigned short* __restrict__ hB,          // pong
                    const float* __restrict__ x,              // (H, T)
                    const float* __restrict__ Whx,            // (H)
                    const float* __restrict__ biasH,          // (H)
                    unsigned int* __restrict__ cnt,           // gridDim.y * T counters (zeroed)
                    int H, int T)
{
    __shared__ unsigned short lA[2][TILE_M][LDSK];
    __shared__ unsigned short lB[2][TILE_N][LDSK];

    const int tid  = threadIdx.x;
    const int lane = tid & 31;
    const int wave = tid >> 5;
    const int hl   = lane >> 4;
    const int ln   = lane & 15;

    const int mBase   = blockIdx.y * TILE_M;
    const int nBase   = blockIdx.x * TILE_N;
    const int rowBase = mBase + wave * 16 + (hl ? 8 : 0);
    const unsigned int nArrive = gridDim.x;        // 8 blocks per stripe

    // invariants of the t-loop
    float wrow[8], bc[8];
    #pragma unroll
    for (int r = 0; r < 8; ++r) wrow[r] = Whx[rowBase + r];
    #pragma unroll
    for (int nt = 0; nt < 8; ++nt) bc[nt] = biasH[nBase + nt * 16 + ln];

    for (int t = 0; t < T; ++t) {
        const unsigned short* src = (t & 1) ? hB : hA;
        unsigned short*       dst = (t & 1) ? hA : hB;

        v8f acc[8];
        mm_core(src + (size_t)mBase * H, WhhT + (size_t)nBase * H, H,
                lA, lB, tid, wave, hl, ln, acc);

        // epilogue: h = tanh(acc + outer(Whx, x_t) + bias_h), bf16
        #pragma unroll
        for (int nt = 0; nt < 8; ++nt) {
            const int col = nBase + nt * 16 + ln;
            const float xv = x[(size_t)col * T + t];
            #pragma unroll
            for (int r = 0; r < 8; ++r) {
                float v = acc[nt][r] + bc[nt] + wrow[r] * xv;
                dst[(size_t)(rowBase + r) * H + col] = f32_to_bf16(tanhf(v));
            }
        }

        // stripe barrier: release stores device-wide, arrive, acquire-spin to 8
        __threadfence();                           // each thread flushes its h stores
        __syncthreads();
        if (tid == 0) {
            unsigned int* c = &cnt[blockIdx.y * T + t];
            __hip_atomic_fetch_add(c, 1u, __ATOMIC_RELEASE, __HIP_MEMORY_SCOPE_AGENT);
            while (__hip_atomic_load(c, __ATOMIC_ACQUIRE, __HIP_MEMORY_SCOPE_AGENT) < nArrive)
                __builtin_amdgcn_s_sleep(1);
        }
        __syncthreads();                           // whole block sees fresh stripe
    }
}

// p = h_T @ W_ph + bias_p  (f32 out)
__global__ __launch_bounds__(THREADS)
void classifier_gemm(const unsigned short* __restrict__ A,     // h_T, H x K bf16
                     const unsigned short* __restrict__ WphT,  // C x K bf16 (n-major)
                     float* __restrict__ P,
                     const float* __restrict__ biasP,
                     int K, int ldc)
{
    __shared__ unsigned short lA[2][TILE_M][LDSK];
    __shared__ unsigned short lB[2][TILE_N][LDSK];

    const int tid  = threadIdx.x;
    const int lane = tid & 31;
    const int wave = tid >> 5;
    const int hl   = lane >> 4;
    const int ln   = lane & 15;

    const int mBase = blockIdx.y * TILE_M;
    const int nBase = blockIdx.x * TILE_N;

    v8f acc[8];
    mm_core(A + (size_t)mBase * K, WphT + (size_t)nBase * K, K,
            lA, lB, tid, wave, hl, ln, acc);

    const int rowBase = mBase + wave * 16 + (hl ? 8 : 0);
    #pragma unroll
    for (int nt = 0; nt < 8; ++nt) {
        const int col = nBase + nt * 16 + ln;
        const float bcv = biasP[col];
        #pragma unroll
        for (int r = 0; r < 8; ++r)
            P[(size_t)(rowBase + r) * ldc + col] = acc[nt][r] + bcv;
    }
}

// One-time: dst[n*K + k] = bf16(src[k*N + n])   (transpose + downconvert weights)
__global__ __launch_bounds__(256)
void transpose_to_bf16(const float* __restrict__ src, unsigned short* __restrict__ dst,
                       int K, int N) {
    int idx = blockIdx.x * 256 + threadIdx.x;
    if (idx < K * N) {
        int n = idx % N;                 // coalesced read over n
        int k = idx / N;
        dst[(size_t)n * K + k] = f32_to_bf16(src[idx]);
    }
}

__global__ __launch_bounds__(256) void zero_u16(unsigned short* __restrict__ p, int n) {
    int i = blockIdx.x * 256 + threadIdx.x;
    if (i < n) p[i] = 0;
}

__global__ __launch_bounds__(256) void zero_u32(unsigned int* __restrict__ p, int n) {
    int i = blockIdx.x * 256 + threadIdx.x;
    if (i < n) p[i] = 0u;
}

// In-place numerically-stable softmax over each row of P (rows = gridDim.x).
__global__ __launch_bounds__(256) void softmax_rows(float* __restrict__ P, int C) {
    __shared__ float red[256];
    const int tid = threadIdx.x;
    float* p = P + (size_t)blockIdx.x * C;

    float m = -3.402823466e38f;
    for (int c = tid; c < C; c += 256) m = fmaxf(m, p[c]);
    red[tid] = m; __syncthreads();
    #pragma unroll
    for (int s = 128; s > 0; s >>= 1) {
        if (tid < s) red[tid] = fmaxf(red[tid], red[tid + s]);
        __syncthreads();
    }
    m = red[0]; __syncthreads();

    float sum = 0.0f;
    for (int c = tid; c < C; c += 256) {
        float e = expf(p[c] - m);
        p[c] = e;
        sum += e;
    }
    red[tid] = sum; __syncthreads();
    #pragma unroll
    for (int s = 128; s > 0; s >>= 1) {
        if (tid < s) red[tid] += red[tid + s];
        __syncthreads();
    }
    const float inv = 1.0f / red[0];
    for (int c = tid; c < C; c += 256) p[c] *= inv;
}

extern "C" void kernel_launch(void* const* d_in, const int* in_sizes, int n_in,
                              void* d_out, int out_size, void* d_ws, size_t ws_size,
                              hipStream_t stream) {
    (void)in_sizes; (void)n_in; (void)out_size; (void)ws_size;
    const int H = 1024, T = 256, C = 512;

    const float* x     = (const float*)d_in[0];  // (H, T)
    const float* Whx   = (const float*)d_in[1];  // (H, 1)
    const float* Whh   = (const float*)d_in[2];  // (H, H)
    const float* biasH = (const float*)d_in[3];  // (H)
    const float* Wph   = (const float*)d_in[4];  // (H, C)
    const float* biasP = (const float*)d_in[5];  // (C)
    float* out = (float*)d_out;                  // (H, C)

    // Workspace: bf16 h ping/pong + transposed bf16 weights + stripe barrier counters.
    unsigned short* hA   = (unsigned short*)d_ws;        // 2 MB
    unsigned short* hB   = hA + (size_t)H * H;           // 2 MB
    unsigned short* WhhT = hB + (size_t)H * H;           // 2 MB  (H x H, n-major)
    unsigned short* WphT = WhhT + (size_t)H * H;         // 1 MB  (C x H, n-major)
    unsigned int*   cnt  = (unsigned int*)(WphT + (size_t)C * H);  // 8 * T u32

    const int nStripes = H / TILE_M;                     // 8

    // One-time prep: transposed bf16 weights, zeroed h0 and barrier counters.
    transpose_to_bf16<<<(H * H + 255) / 256, 256, 0, stream>>>(Whh, WhhT, H, H);
    transpose_to_bf16<<<(H * C + 255) / 256, 256, 0, stream>>>(Wph, WphT, H, C);
    zero_u16<<<(H * H + 255) / 256, 256, 0, stream>>>(hA, H * H);
    zero_u32<<<(nStripes * T + 255) / 256, 256, 0, stream>>>(cnt, nStripes * T);

    // Entire 256-step recurrence in one persistent kernel (8x8 blocks, stripe barriers).
    dim3 gridR(H / TILE_N, nStripes);        // (8, 8)
    rnn_persistent<<<gridR, dim3(THREADS), 0, stream>>>(WhhT, hA, hB, x, Whx, biasH,
                                                        cnt, H, T);
    // T even -> final h in hA.

    // p = h_T @ W_ph + bias_p -> d_out (f32)
    dim3 gridF(C / TILE_N, H / TILE_M);      // (4, 8)
    classifier_gemm<<<gridF, dim3(THREADS), 0, stream>>>(hA, WphT, out, biasP, H, C);

    // softmax over class axis, in place on d_out
    softmax_rows<<<H, 256, 0, stream>>>(out, C);
}
